// UnpNet_11467562680757
// MI455X (gfx1250) — compile-verified, matrix operand order checked by reading, not run on hardware
//
#include <hip/hip_runtime.h>

typedef _Float16 v16h __attribute__((ext_vector_type(16)));
typedef _Float16 h8   __attribute__((ext_vector_type(8)));
typedef _Float16 h4   __attribute__((ext_vector_type(4)));
typedef float    v8f  __attribute__((ext_vector_type(8)));

#define BB   2
#define CIN  64
#define CC   128
#define HH   64
#define WW   1024
#define HY2  32
#define WX2  512
#define HWF  (HH * WW)     // 65536
#define HW2  (HY2 * WX2)   // 16384

__device__ __forceinline__ float actf(float v, int mode) {
  if (mode == 1) return v > 0.f ? v : 0.01f * v;   // leaky_relu (jax default slope 0.01)
  if (mode == 2) return v > 0.f ? v : 0.f;         // relu
  return v;
}

__device__ __forceinline__ v16h cat16(h8 lo, h8 hi) {
  return __builtin_shufflevector(lo, hi, 0,1,2,3,4,5,6,7,8,9,10,11,12,13,14,15);
}

// ---------------------------------------------------------------------------
// NCHW f32 -> NHWC f16 pack of x
// ---------------------------------------------------------------------------
__global__ void pack_x_kernel(const float* __restrict__ x, _Float16* __restrict__ xh) {
  int idx = blockIdx.x * 256 + threadIdx.x;
  if (idx >= BB * HWF * CIN) return;
  int c   = idx % CIN;
  int pix = idx / CIN;
  int xx  = pix % WW;
  int yy  = (pix / WW) % HH;
  int b   = pix / (WW * HH);
  xh[idx] = (_Float16)x[((size_t)(b * CIN + c) * HH + yy) * WW + xx];
}

// ---------------------------------------------------------------------------
// Pack weights into WMMA B-fragment layout:
//   out[((tap*ksteps+ks)*8 + nt)*512 + lane*16 + j]
//   lane<16:  n = nt*16+lane,     j<8 -> k = k0+j,     j>=8 -> k = k0+16+(j-8)
//   lane>=16: n = nt*16+lane-16,  j<8 -> k = k0+8+j,   j>=8 -> k = k0+24+(j-8)
// mode 0: w[n*Ktap + k]   (1x1 / linear, [N,K])
// mode 1: w[(n*Ktap + k)*taps + tap]   (conv OIHW, k = input channel)
// ---------------------------------------------------------------------------
__global__ void pack_w_kernel(const float* __restrict__ w, _Float16* __restrict__ out,
                              int Ktap, int taps, int mode) {
  int total = taps * Ktap * 128;
  int idx = blockIdx.x * 256 + threadIdx.x;
  if (idx >= total) return;
  int j    = idx & 15;
  int lane = (idx >> 4) & 31;
  int nt   = (idx >> 9) & 7;
  int rest = idx >> 12;               // tap*ksteps + ks
  int ksteps = Ktap >> 5;
  int ks  = rest % ksteps;
  int tap = rest / ksteps;
  int n  = nt * 16 + (lane & 15);
  int kk = ks * 32 + ((lane >> 4) ? 8 : 0) + ((j >> 3) ? 16 : 0) + (j & 7);
  float v;
  if (mode == 0) v = w[(size_t)n * Ktap + kk];
  else           v = w[((size_t)n * Ktap + kk) * taps + tap];
  out[idx] = (_Float16)v;
}

// ---------------------------------------------------------------------------
// Generic implicit-GEMM conv with double-buffered async global->LDS staging.
// Tile: 128 consecutive pixels (one row) x 128 output channels. 8 waves,
// wave w -> pixel rows [16w,16w+16), all 8 N-tiles; WMMA f32_16x16x32_f16.
// A tiles are moved with GLOBAL_LOAD_ASYNC_TO_LDS_B128 (ASYNCcnt), overlapped
// with the WMMA chain on the other LDS buffer; one barrier per K-step.
// ---------------------------------------------------------------------------
__global__ __launch_bounds__(256)
void gemm_conv(const _Float16* __restrict__ in, int inPixStride,
               const _Float16* __restrict__ wp,
               const float* __restrict__ bias,
               const float* __restrict__ scale,
               const float* __restrict__ shift,
               const _Float16* __restrict__ addsrc,          // f16, pixStride 128
               _Float16* __restrict__ out_h, int outPixStride, int outChanOff,
               float* __restrict__ out_f,                    // NCHW f32 (nullable)
               int Hh, int Ww, int Ktap, int tapsY, int tapsX, int dil, int off,
               int actMode, int actAfterBn)
{
  __shared__ _Float16 As[2][128 * 40];       // double-buffered 128 px x 32 K
  const int tid  = threadIdx.x;
  const int lane = tid & 31;
  const int wv   = tid >> 5;
  const int tilesX = Ww >> 7;
  const int id  = blockIdx.x;
  const int x0  = (id % tilesX) << 7;
  const int y   = (id / tilesX) % Hh;
  const int bi  = id / (tilesX * Hh);
  const int ksteps = Ktap >> 5;
  const int taps   = tapsY * tapsX;
  const int T      = taps * ksteps;

  v8f acc[8] = {};

  const int pi   = tid >> 1;                 // staging: pixel in tile
  const int part = (tid & 1) << 4;           // staging: half-chunk (0 or 16)

  // Issue async copy of one (tap, kstep) A-slab into LDS buffer `buf`.
  auto stage = [&](int tap, int ks, int buf) {
    const int dy = (tap / tapsX) * dil + off;
    const int dx = (tap % tapsX) * dil + off;
    const int yy = y + dy;
    const int xx = x0 + pi + dx;
    _Float16* dst = &As[buf][pi * 40 + part];
    if (yy >= 0 && yy < Hh && xx >= 0 && xx < Ww) {
      const _Float16* src = in + ((size_t)(bi * Hh + yy) * Ww + xx) * inPixStride
                               + (ks << 5) + part;
      unsigned l0 = (unsigned)(uintptr_t)dst;   // LDS byte offset (flat addr[31:0])
      asm volatile("global_load_async_to_lds_b128 %0, %1, off\n\t"
                   "global_load_async_to_lds_b128 %0, %1, off offset:16"
                   :: "v"(l0), "v"(src) : "memory");
    } else {
      h8 z = {};
      *(h8*)dst = z; *(h8*)(dst + 8) = z;     // compiler-tracked, fenced by barrier
    }
  };

  // prologue: stage step 0 into buffer 0
  stage(0, 0, 0);
  int tapN = 0, ksN = 1;                      // next (tap, ks) to stage
  if (ksN == ksteps) { ksN = 0; tapN = 1; }

  const int nlo = lane & 15;
  const int kb  = (lane >> 4) << 3;

  for (int s = 0; s < T; ++s) {
    const int cur = s & 1;
    // own async copies into As[cur] complete, then publish to the workgroup
    asm volatile("s_wait_asynccnt 0x0" ::: "memory");
    __syncthreads();
    // prefetch next slab into the other buffer (everyone is done reading it)
    if (s + 1 < T) {
      stage(tapN, ksN, cur ^ 1);
      if (++ksN == ksteps) { ksN = 0; ++tapN; }
    }
    // ---- A fragment (per-lane layout per CDNA5 ISA 16-bit A 16x32) ----
    const _Float16* ar = &As[cur][(wv * 16 + nlo) * 40 + kb];
    v16h a = cat16(*(const h8*)ar, *(const h8*)(ar + 16));
    // ---- all 8 B fragments first, then back-to-back WMMAs ----
    const _Float16* wbase = wp + (size_t)(s * 8) * 512 + (size_t)lane * 16;
    v16h bfr[8];
    #pragma unroll
    for (int nt = 0; nt < 8; ++nt)
      bfr[nt] = *(const v16h*)(wbase + (size_t)nt * 512);
    #pragma unroll
    for (int nt = 0; nt < 8; ++nt)
      acc[nt] = __builtin_amdgcn_wmma_f32_16x16x32_f16(
          false, a, false, bfr[nt], (short)0, acc[nt], false, false);
  }

  // ---- epilogue: bias -> (act) -> bn -> (act) -> +shortcut -> stores ----
  const int mhi = (lane >> 4) << 3;
  const int r0  = wv << 4;
  #pragma unroll
  for (int nt = 0; nt < 8; ++nt) {
    const int c = (nt << 4) + nlo;
    const float bsv = bias  ? bias[c]  : 0.f;
    const float sv  = scale ? scale[c] : 1.f;
    const float shv = shift ? shift[c] : 0.f;
    #pragma unroll
    for (int r = 0; r < 8; ++r) {
      float v = acc[nt][r] + bsv;
      if (!actAfterBn) v = actf(v, actMode);
      v = v * sv + shv;
      if (actAfterBn) v = actf(v, actMode);
      const int xl = x0 + r0 + r + mhi;
      const size_t pix = (size_t)(bi * Hh + y) * Ww + xl;
      if (addsrc) v += (float)addsrc[pix * 128 + c];
      if (out_h) out_h[pix * outPixStride + outChanOff + c] = (_Float16)v;
      if (out_f) out_f[(((size_t)bi * 128 + c) * Hh + y) * Ww + xl] = v;
    }
  }
}

// ---------------------------------------------------------------------------
// gxyz geometry + 3-layer WeightNet chain (3->8->8->16), fused.
// One thread per (batch, out-pixel, tap k). Pad: g = 0 - center at OOB.
// Output layout: wn[(pixel*25 + k)*16 + n]  (f16)
// ---------------------------------------------------------------------------
__global__ void wn_chain(const float* __restrict__ xyz,
                         const float* __restrict__ w1, const float* __restrict__ b1,
                         const float* __restrict__ w2, const float* __restrict__ b2,
                         const float* __restrict__ w3, const float* __restrict__ b3,
                         const float* __restrict__ s1, const float* __restrict__ h1,
                         const float* __restrict__ s2, const float* __restrict__ h2,
                         const float* __restrict__ s3, const float* __restrict__ h3,
                         _Float16* __restrict__ wnb)
{
  int idx = blockIdx.x * 256 + threadIdx.x;
  if (idx >= BB * HW2 * 25) return;
  int k  = idx % 25;
  int p  = idx / 25;
  int x2 = p % WX2;
  int y2 = (p / WX2) % HY2;
  int b  = p / (WX2 * HY2);
  int cy = y2 * 2, cx = x2 * 2;
  int ry = cy + k / 5 - 2, rx = cx + k % 5 - 2;
  bool inb = (ry >= 0 && ry < HH && rx >= 0 && rx < WW);
  float g[3];
  #pragma unroll
  for (int c = 0; c < 3; ++c) {
    float ctr = xyz[((size_t)(b * 3 + c) * HH + cy) * WW + cx];
    float v   = inb ? xyz[((size_t)(b * 3 + c) * HH + ry) * WW + rx] : 0.f;
    g[c] = v - ctr;
  }
  float t1[8];
  #pragma unroll
  for (int o = 0; o < 8; ++o) {
    float v = b1[o];
    #pragma unroll
    for (int c = 0; c < 3; ++c) v += w1[o * 3 + c] * g[c];
    v = v * s1[o] + h1[o];
    t1[o] = v > 0.f ? v : 0.f;
  }
  float t2[8];
  #pragma unroll
  for (int o = 0; o < 8; ++o) {
    float v = b2[o];
    #pragma unroll
    for (int c = 0; c < 8; ++c) v += w2[o * 8 + c] * t1[c];
    v = v * s2[o] + h2[o];
    t2[o] = v > 0.f ? v : 0.f;
  }
  #pragma unroll
  for (int o = 0; o < 16; ++o) {
    float v = b3[o];
    #pragma unroll
    for (int c = 0; c < 8; ++c) v += w3[o * 8 + c] * t2[c];
    v = v * s3[o] + h3[o];
    v = v > 0.f ? v : 0.f;
    wnb[((size_t)p * 25 + k) * 16 + o] = (_Float16)v;
  }
}

// ---------------------------------------------------------------------------
// Per-pixel einsum  Y[c,n] = sum_k unfold(h)[c,k] * wn[n,k]  via WMMA.
// One wave per output pixel; U (128x25, pad K->32) staged in LDS;
// Wn is the B fragment (25x16 pad 32x16). Writes ybuf[pixel][c*16+n].
// ---------------------------------------------------------------------------
__global__ __launch_bounds__(128)
void pconv_e(const _Float16* __restrict__ hb,
             const _Float16* __restrict__ wnb,
             _Float16* __restrict__ yb)
{
  __shared__ _Float16 Usm[4 * 128 * 40];     // 40 KB
  const int lane = threadIdx.x & 31;
  const int wv   = threadIdx.x >> 5;
  const int pixel = blockIdx.x * 4 + wv;
  const int x2 = pixel % WX2;
  const int y2 = (pixel / WX2) % HY2;
  const int b  = pixel / (WX2 * HY2);
  _Float16* U = &Usm[wv * (128 * 40)];
  const int c0 = lane * 4;
  for (int k = 0; k < 25; ++k) {
    int ry = y2 * 2 + k / 5 - 2;
    int rx = x2 * 2 + k % 5 - 2;
    h4 v = {};
    if (ry >= 0 && ry < HH && rx >= 0 && rx < WW)
      v = *(const h4*)&hb[((size_t)(b * HH + ry) * WW + rx) * 128 + c0];
    U[(c0 + 0) * 40 + k] = v[0];
    U[(c0 + 1) * 40 + k] = v[1];
    U[(c0 + 2) * 40 + k] = v[2];
    U[(c0 + 3) * 40 + k] = v[3];
  }
  for (int k = 25; k < 32; ++k)
    #pragma unroll
    for (int q = 0; q < 4; ++q)
      U[(c0 + q) * 40 + k] = (_Float16)0.f;
  __syncthreads();

  const int n  = lane & 15;
  const int kb = (lane >> 4) << 3;
  v16h bf;
  #pragma unroll
  for (int j = 0; j < 8; ++j) {
    int k = kb + j;
    bf[j] = (k < 25) ? wnb[((size_t)pixel * 25 + k) * 16 + n] : (_Float16)0.f;
  }
  #pragma unroll
  for (int j = 0; j < 8; ++j) {
    int k = kb + 16 + j;
    bf[j + 8] = (k < 25) ? wnb[((size_t)pixel * 25 + k) * 16 + n] : (_Float16)0.f;
  }
  const int mhi = (lane >> 4) << 3;
  #pragma unroll
  for (int mt = 0; mt < 8; ++mt) {
    const _Float16* ar = &U[(mt * 16 + n) * 40 + kb];
    v16h a = cat16(*(const h8*)ar, *(const h8*)(ar + 16));
    v8f accv = {};
    accv = __builtin_amdgcn_wmma_f32_16x16x32_f16(
        false, a, false, bf, (short)0, accv, false, false);
    #pragma unroll
    for (int r = 0; r < 8; ++r) {
      int cch = mt * 16 + r + mhi;
      yb[(size_t)pixel * 2048 + cch * 16 + n] = (_Float16)accv[r];
    }
  }
}

// ---------------------------------------------------------------------------
extern "C" void kernel_launch(void* const* d_in, const int* in_sizes, int n_in,
                              void* d_out, int out_size, void* d_ws, size_t ws_size,
                              hipStream_t stream) {
  const float* x      = (const float*)d_in[0];
  const float* xyz    = (const float*)d_in[1];
  const float* c1_w   = (const float*)d_in[2];
  const float* c1_b   = (const float*)d_in[3];
  const float* c2_w   = (const float*)d_in[4];
  const float* c2_b   = (const float*)d_in[5];
  const float* c3_w   = (const float*)d_in[6];
  const float* c3_b   = (const float*)d_in[7];
  const float* c4_w   = (const float*)d_in[8];
  const float* c4_b   = (const float*)d_in[9];
  const float* c5_w   = (const float*)d_in[10];
  const float* c5_b   = (const float*)d_in[11];
  const float* rbn_s  = (const float*)d_in[12];
  const float* rbn_b  = (const float*)d_in[13];
  const float* p_w    = (const float*)d_in[14];
  const float* p_b    = (const float*)d_in[15];
  const float* pbn_s  = (const float*)d_in[16];
  const float* pbn_b  = (const float*)d_in[17];
  const float* lin_w  = (const float*)d_in[18];
  const float* lin_b  = (const float*)d_in[19];
  const float* w1_w   = (const float*)d_in[20];
  const float* w1_b   = (const float*)d_in[21];
  const float* w2_w   = (const float*)d_in[22];
  const float* w2_b   = (const float*)d_in[23];
  const float* w3_w   = (const float*)d_in[24];
  const float* w3_b   = (const float*)d_in[25];
  const float* wbn1_s = (const float*)d_in[26];
  const float* wbn1_b = (const float*)d_in[27];
  const float* wbn2_s = (const float*)d_in[28];
  const float* wbn2_b = (const float*)d_in[29];
  const float* wbn3_s = (const float*)d_in[30];
  const float* wbn3_b = (const float*)d_in[31];

  float* outF = (float*)d_out;
  const size_t RESB_N = (size_t)BB * CC * HY2 * WX2;   // 4,194,304 floats

  // workspace layout (bytes); ybuf aliases the dead xh/sc/cat region
  char* ws = (char*)d_ws;
  _Float16* xh    = (_Float16*)(ws + 0);                 //  16.8 MB
  _Float16* sc    = (_Float16*)(ws + 16777216);          //  33.6 MB
  _Float16* cat   = (_Float16*)(ws + 50331648);          // 100.7 MB
  _Float16* tmpA  = cat;                                 // alias (cat dead after c5)
  _Float16* tmpB  = (_Float16*)(ws + 50331648 + 33554432);
  _Float16* ybuf  = (_Float16*)(ws + 0);                 // alias (region dead by E)
  _Float16* resAh = (_Float16*)(ws + 150994944);         //  33.6 MB
  _Float16* hbuf  = (_Float16*)(ws + 184549376);         //  33.6 MB
  _Float16* wnb   = (_Float16*)(ws + 218103808);         //  26.2 MB
  _Float16* wpk   = (_Float16*)(ws + 244318208);         //   1.3 MB packed weights
  _Float16* pk_c1  = wpk + 0;        // 1*64*128
  _Float16* pk_c2  = wpk + 8192;     // 9*64*128
  _Float16* pk_c3  = wpk + 81920;    // 9*128*128
  _Float16* pk_c4  = wpk + 229376;   // 4*128*128
  _Float16* pk_c5  = wpk + 294912;   // 1*384*128
  _Float16* pk_p0  = wpk + 344064;   // 1*128*128
  _Float16* pk_p1  = wpk + 360448;
  _Float16* pk_p2  = wpk + 376832;
  _Float16* pk_lin = wpk + 393216;   // 1*2048*128

  const int gFull = BB * HH * (WW >> 7);    // 1024 blocks
  const int gHalf = BB * HY2 * (WX2 >> 7);  //  256 blocks

  // ---- pack inputs and weights ----
  pack_x_kernel<<<(BB * HWF * CIN + 255) / 256, 256, 0, stream>>>(x, xh);
  auto PW = [&](const float* w, _Float16* o, int Ktap, int taps, int mode) {
    int tot = taps * Ktap * 128;
    pack_w_kernel<<<(tot + 255) / 256, 256, 0, stream>>>(w, o, Ktap, taps, mode);
  };
  PW(c1_w, pk_c1, 64, 1, 0);
  PW(c2_w, pk_c2, 64, 9, 1);
  PW(c3_w, pk_c3, 128, 9, 1);
  PW(c4_w, pk_c4, 128, 4, 1);
  PW(c5_w, pk_c5, 384, 1, 0);
  PW(p_w + 0 * 16384, pk_p0, 128, 1, 0);
  PW(p_w + 1 * 16384, pk_p1, 128, 1, 0);
  PW(p_w + 2 * 16384, pk_p2, 128, 1, 0);
  PW(lin_w, pk_lin, 2048, 1, 0);

  // ---- shortcut = lrelu(c1x1(x)) ----
  gemm_conv<<<gFull, 256, 0, stream>>>(xh, CIN, pk_c1, c1_b, nullptr, nullptr,
      nullptr, sc, 128, 0, nullptr, HH, WW, 64, 1, 1, 1, 0, 1, 0);
  // ---- resA1 = bn0(lrelu(conv3x3)) -> cat[:,0:128] ----
  gemm_conv<<<gFull, 256, 0, stream>>>(xh, CIN, pk_c2, c2_b, rbn_s, rbn_b,
      nullptr, cat, 384, 0, nullptr, HH, WW, 64, 3, 3, 1, -1, 1, 0);
  // ---- resA2 = bn1(lrelu(conv3x3 dil2)) -> cat[:,128:256] ----
  gemm_conv<<<gFull, 256, 0, stream>>>(cat, 384, pk_c3, c3_b, rbn_s + 128, rbn_b + 128,
      nullptr, cat, 384, 128, nullptr, HH, WW, 128, 3, 3, 2, -2, 1, 0);
  // ---- resA3 = bn2(lrelu(conv2x2 dil2)) -> cat[:,256:384] ----
  gemm_conv<<<gFull, 256, 0, stream>>>(cat + 128, 384, pk_c4, c4_b, rbn_s + 256, rbn_b + 256,
      nullptr, cat, 384, 256, nullptr, HH, WW, 128, 2, 2, 2, -1, 1, 0);
  // ---- resA = bn3(lrelu(c5(cat))) + shortcut ; f32 out + f16 copy ----
  gemm_conv<<<gFull, 256, 0, stream>>>(cat, 384, pk_c5, c5_b, rbn_s + 384, rbn_b + 384,
      sc, resAh, 128, 0, outF + RESB_N, HH, WW, 384, 1, 1, 1, 0, 1, 0);
  // ---- p-chain: relu(bn(c1x1)) x3 ----
  gemm_conv<<<gFull, 256, 0, stream>>>(resAh, 128, pk_p0, p_b, pbn_s, pbn_b,
      nullptr, tmpA, 128, 0, nullptr, HH, WW, 128, 1, 1, 1, 0, 2, 1);
  gemm_conv<<<gFull, 256, 0, stream>>>(tmpA, 128, pk_p1, p_b + 128, pbn_s + 128, pbn_b + 128,
      nullptr, tmpB, 128, 0, nullptr, HH, WW, 128, 1, 1, 1, 0, 2, 1);
  gemm_conv<<<gFull, 256, 0, stream>>>(tmpB, 128, pk_p2, p_b + 256, pbn_s + 256, pbn_b + 256,
      nullptr, hbuf, 128, 0, nullptr, HH, WW, 128, 1, 1, 1, 0, 2, 1);
  // ---- WeightNet chain on geometry ----
  wn_chain<<<(BB * HW2 * 25 + 255) / 256, 256, 0, stream>>>(xyz,
      w1_w, w1_b, w2_w, w2_b, w3_w, w3_b,
      wbn1_s, wbn1_b, wbn2_s, wbn2_b, wbn3_s, wbn3_b, wnb);
  // ---- per-pixel U x Wn -> ybuf [pixel][2048] ----
  pconv_e<<<(BB * HW2) / 4, 128, 0, stream>>>(hbuf, wnb, ybuf);
  // ---- resB = relu(bn3(lin(ybuf))) -> f32 NCHW output ----
  gemm_conv<<<gHalf, 256, 0, stream>>>(ybuf, 2048, pk_lin, lin_b, pbn_s + 384, pbn_b + 384,
      nullptr, nullptr, 0, 0, outF, HY2, WX2, 2048, 1, 1, 1, 0, 2, 1);
}
